// ClassifierGAT_39934605918642
// MI455X (gfx1250) — matrix operand order repaired and checked by reading, not compiled, not run on hardware
//
#include <hip/hip_runtime.h>
#include <math.h>

// ---------------- problem constants (match reference) ----------------
#define N_NODES 20000
#define N_EDGES 320000
#define IN_DIM  512
#define HID     64
#define NH      8
#define NHID    (NH * HID)   // 512
#define NCLS    20
#define NG      64
#define BN_EPS  1e-5f
#define NEG_BIG (-1.0e30f)
#define MT      5            // M-tiles (16 rows each) per wave; 20000 = 80*250

typedef __attribute__((ext_vector_type(16))) __bf16 v16bf;
typedef __attribute__((ext_vector_type(8)))  float  v8f;

// ---------------------------------------------------------------------
// Pre-pack fp32 weight B[K,N] -> bf16 in the B-fragment layout used by the GEMM below (contiguous k order):
//   Bp[tn][c][lane][j], lane 0-15: col=tn*16+lane, K = c*32 + j (j=0..15)
//                       lane 16-31: col=tn*16+lane-16, K = c*32 + 16 + j
// One contiguous 32B run per (tile, chunk, lane).
// ---------------------------------------------------------------------
__global__ void pack_weight_bf16(const float* __restrict__ B, __bf16* __restrict__ Bp,
                                 int N, int K)
{
    int t = blockIdx.x * blockDim.x + threadIdx.x;
    int chunks = K >> 5;
    int total  = (N >> 4) * chunks * 512;      // 32 lanes * 16 values
    if (t >= total) return;
    int j    = t & 15;
    int lane = (t >> 4) & 31;
    int c    = (t >> 9) % chunks;
    int tn   = (t >> 9) / chunks;
    int n = tn * 16 + (lane & 15);
    int k = c * 32 + (lane >> 4) * 16 + j;
    Bp[t] = (__bf16)B[(size_t)k * N + n];
}

__device__ __forceinline__ v16bf cvt_frag(const float* p)
{
    const float4 f0 = *(const float4*)(p);
    const float4 f1 = *(const float4*)(p + 4);
    const float4 f2 = *(const float4*)(p + 8);
    const float4 f3 = *(const float4*)(p + 12);
    v16bf a;
    a[0]  = (__bf16)f0.x; a[1]  = (__bf16)f0.y;
    a[2]  = (__bf16)f0.z; a[3]  = (__bf16)f0.w;
    a[4]  = (__bf16)f1.x; a[5]  = (__bf16)f1.y;
    a[6]  = (__bf16)f1.z; a[7]  = (__bf16)f1.w;
    a[8]  = (__bf16)f2.x; a[9]  = (__bf16)f2.y;
    a[10] = (__bf16)f2.z; a[11] = (__bf16)f2.w;
    a[12] = (__bf16)f3.x; a[13] = (__bf16)f3.y;
    a[14] = (__bf16)f3.z; a[15] = (__bf16)f3.w;
    return a;
}

// ---------------------------------------------------------------------
// WMMA GEMM: C[M,N] = A[M,K] @ B[K,N] (+bias[N]); A fp32 (converted in-regs),
// B pre-packed bf16.  One wave computes an 80x16 C block (5 WMMA tiles
// sharing each B fragment).  M%80==0, N%16==0, K%32==0.
// Per K-chunk: build all 5 A fragments in distinct registers first, then
// issue the 5 WMMAs back-to-back -> only one WMMA->VALU WAR hazard per
// chunk instead of five (ISA 7.12.1: 4 coexec NOPs each).
// A-fragment layout (matches packed B): lanes 0-15 row=l, K=k0..k0+15;
// lanes 16-31 row=l-16, K=k0+16..k0+31.
// C/D: vgpr r -> row half*8+r, lane -> col.
// ---------------------------------------------------------------------
__global__ __launch_bounds__(128)
void wmma_gemm_packed(const float* __restrict__ A, const __bf16* __restrict__ Bp,
                      const float* __restrict__ bias, float* __restrict__ C,
                      int M, int N, int K)
{
    const int lane   = threadIdx.x & 31;
    const int wave   = threadIdx.x >> 5;
    const int tilesN = N >> 4;
    const int waves  = (M / (16 * MT)) * tilesN;
    const int w      = blockIdx.x * 4 + wave;
    if (w >= waves) return;                 // wave-uniform: EXEC stays all-1s
    const int bm = (w / tilesN) * (16 * MT);
    const int tn = w % tilesN;

    const int half   = lane >> 4;
    const int l15    = lane & 15;
    const int chunks = K >> 5;

    const v16bf* bptr = (const v16bf*)Bp + (size_t)tn * chunks * 32 + lane;

    const float* arow[MT];
#pragma unroll
    for (int i = 0; i < MT; ++i)
        arow[i] = A + (size_t)(bm + i * 16 + l15) * K + half * 16;

    const v8f vzero = {};
    v8f acc[MT];
#pragma unroll
    for (int i = 0; i < MT; ++i) acc[i] = vzero;

    for (int c = 0; c < chunks; ++c) {
        const v16bf b = bptr[(size_t)c * 32];
        v16bf afr[MT];
#pragma unroll
        for (int i = 0; i < MT; ++i)
            afr[i] = cvt_frag(arow[i] + c * 32);
#pragma unroll
        for (int i = 0; i < MT; ++i)
            acc[i] = __builtin_amdgcn_wmma_f32_16x16x32_bf16(
                         false, afr[i], false, b, (short)0, acc[i], false, false);
    }

    const float bv = bias ? bias[tn * 16 + l15] : 0.0f;
#pragma unroll
    for (int i = 0; i < MT; ++i) {
        float* crow = C + (size_t)(bm + i * 16 + half * 8) * N + tn * 16 + l15;
#pragma unroll
        for (int r = 0; r < 8; ++r)
            crow[(size_t)r * N] = acc[i][r] + bv;
    }
}

// ---------------------------------------------------------------------
__device__ inline void atomicMaxF(float* addr, float v)
{
    unsigned int* ai  = (unsigned int*)addr;
    unsigned int  old = *ai, assumed;
    do {
        assumed = old;
        if (__uint_as_float(assumed) >= v) break;
        old = atomicCAS(ai, assumed, __float_as_uint(v));
    } while (old != assumed);
}

__global__ void fill_f32(float* p, float v, int n)
{
    int t = blockIdx.x * blockDim.x + threadIdx.x;
    if (t < n) p[t] = v;
}

// el[i,h] = sum_d feat[i,h,d]*attn_l[h,d] ; er likewise
__global__ void attn_scores(const float* __restrict__ feat,
                            const float* __restrict__ al_,
                            const float* __restrict__ ar_,
                            float* __restrict__ el, float* __restrict__ er)
{
    int t = blockIdx.x * blockDim.x + threadIdx.x;
    if (t >= N_NODES * NH) return;
    int i = t / NH, h = t % NH;
    const float* f  = feat + (size_t)i * NHID + h * HID;
    const float* al = al_ + h * HID;
    const float* ar = ar_ + h * HID;
    float sl = 0.f, sr = 0.f;
#pragma unroll 4
    for (int d = 0; d < HID; ++d) { float v = f[d]; sl += v * al[d]; sr += v * ar[d]; }
    el[t] = sl; er[t] = sr;
}

// e = leaky_relu(el[src]+er[dst], 0.2); running segment max over dst
__global__ void edge_score_max(const int* __restrict__ src, const int* __restrict__ dst,
                               const float* __restrict__ el, const float* __restrict__ er,
                               float* __restrict__ esc, float* __restrict__ m)
{
    int t = blockIdx.x * blockDim.x + threadIdx.x;
    if (t >= N_EDGES * NH) return;
    int e = t / NH, h = t % NH;
    float s = el[src[e] * NH + h] + er[dst[e] * NH + h];
    s = (s > 0.f) ? s : 0.2f * s;
    esc[t] = s;
    atomicMaxF(&m[dst[e] * NH + h], s);
}

__global__ void edge_exp_sum(const int* __restrict__ dst,
                             float* __restrict__ esc,
                             const float* __restrict__ m, float* __restrict__ den)
{
    int t = blockIdx.x * blockDim.x + threadIdx.x;
    if (t >= N_EDGES * NH) return;
    int e = t / NH, h = t % NH;
    float ex = __expf(esc[t] - m[dst[e] * NH + h]);
    esc[t] = ex;
    atomicAdd(&den[dst[e] * NH + h], ex);
}

// esc <- softmax weight alpha (fold denominator in once)
__global__ void edge_alpha(const int* __restrict__ dst,
                           float* __restrict__ esc, const float* __restrict__ den)
{
    int t = blockIdx.x * blockDim.x + threadIdx.x;
    if (t >= N_EDGES * NH) return;
    int e = t / NH, h = t % NH;
    esc[t] = esc[t] / den[dst[e] * NH + h];
}

// thread per (edge, float4 channel): coalesced b128 gather + 4 atomic adds
__global__ void edge_message(const int* __restrict__ src, const int* __restrict__ dst,
                             const float* __restrict__ feat, const float* __restrict__ esc,
                             float* __restrict__ acc)
{
    long long t = (long long)blockIdx.x * blockDim.x + threadIdx.x;
    if (t >= (long long)N_EDGES * (NHID / 4)) return;
    int e = (int)(t >> 7);          // / 128
    int q = (int)(t & 127);         // float4 index; channel c = 4q
    int h = q >> 4;                 // (4q)/64
    float a = esc[e * NH + h];
    const float4 f = *(const float4*)(feat + (size_t)src[e] * NHID + q * 4);
    float* p = acc + (size_t)dst[e] * NHID + q * 4;
    atomicAdd(p + 0, f.x * a);
    atomicAdd(p + 1, f.y * a);
    atomicAdd(p + 2, f.z * a);
    atomicAdd(p + 3, f.w * a);
}

__global__ void leaky_act4(float4* p, int n4)
{
    int t = blockIdx.x * blockDim.x + threadIdx.x;
    if (t >= n4) return;
    float4 v = p[t];
    v.x = (v.x > 0.f) ? v.x : 0.01f * v.x;
    v.y = (v.y > 0.f) ? v.y : 0.01f * v.y;
    v.z = (v.z > 0.f) ? v.z : 0.01f * v.z;
    v.w = (v.w > 0.f) ? v.w : 0.01f * v.w;
    p[t] = v;
}

// batch stats per column of hg = concat([hd, he], axis=1), biased variance
__global__ void bn_stats(const float* __restrict__ hd, const float* __restrict__ he,
                         float* __restrict__ mu, float* __restrict__ var)
{
    int c = blockIdx.x;             // 0..127
    __shared__ float ss[256], sq[256];
    float s = 0.f, q = 0.f;
    for (int i = threadIdx.x; i < N_NODES; i += 256) {
        float v = (c < HID) ? hd[(size_t)i * HID + c] : he[(size_t)i * HID + (c - HID)];
        s += v; q += v * v;
    }
    ss[threadIdx.x] = s; sq[threadIdx.x] = q;
    __syncthreads();
    for (int w = 128; w > 0; w >>= 1) {
        if (threadIdx.x < w) { ss[threadIdx.x] += ss[threadIdx.x + w]; sq[threadIdx.x] += sq[threadIdx.x + w]; }
        __syncthreads();
    }
    if (threadIdx.x == 0) {
        float mean = ss[0] / (float)N_NODES;
        mu[c]  = mean;
        var[c] = sq[0] / (float)N_NODES - mean * mean;
    }
}

__global__ void bn_norm(const float* __restrict__ hd, const float* __restrict__ he,
                        const float* __restrict__ mu, const float* __restrict__ var,
                        const float* __restrict__ gamma, const float* __restrict__ beta,
                        float* __restrict__ hg)
{
    int t = blockIdx.x * blockDim.x + threadIdx.x;
    if (t >= N_NODES * 2 * HID) return;
    int i = t / (2 * HID), c = t % (2 * HID);
    float v = (c < HID) ? hd[(size_t)i * HID + c] : he[(size_t)i * HID + (c - HID)];
    hg[t] = (v - mu[c]) * rsqrtf(var[c] + BN_EPS) * gamma[c] + beta[c];
}

__global__ void gate_proj(const float* __restrict__ hg, const float* __restrict__ gW,
                          const float* __restrict__ gb, float* __restrict__ gate)
{
    int i = blockIdx.x * blockDim.x + threadIdx.x;
    if (i >= N_NODES) return;
    float s = gb[0];
#pragma unroll 4
    for (int c = 0; c < 2 * HID; ++c) s += hg[(size_t)i * 2 * HID + c] * gW[c];
    gate[i] = s;
}

__global__ void seg_gate_max(const int* __restrict__ gid, const float* __restrict__ gate,
                             float* __restrict__ gm)
{
    int i = blockIdx.x * blockDim.x + threadIdx.x;
    if (i < N_NODES) atomicMaxF(&gm[gid[i]], gate[i]);
}

__global__ void seg_gate_expsum(const int* __restrict__ gid, float* __restrict__ gate,
                                const float* __restrict__ gm, float* __restrict__ gden)
{
    int i = blockIdx.x * blockDim.x + threadIdx.x;
    if (i >= N_NODES) return;
    float ge = __expf(gate[i] - gm[gid[i]]);
    gate[i] = ge;
    atomicAdd(&gden[gid[i]], ge);
}

__global__ void pool(const int* __restrict__ gid, const float* __restrict__ gate,
                     const float* __restrict__ gden, const float* __restrict__ hg,
                     float* __restrict__ rdt)
{
    int t = blockIdx.x * blockDim.x + threadIdx.x;
    if (t >= N_NODES * 2 * HID) return;
    int i = t / (2 * HID), c = t % (2 * HID);
    int g = gid[i];
    float alpha = gate[i] / gden[g];
    atomicAdd(&rdt[g * 2 * HID + c], alpha * hg[t]);
}

__global__ void classifier(const float* __restrict__ rdt, const float* __restrict__ clsW,
                           const float* __restrict__ clsb, float* __restrict__ out)
{
    int t = blockIdx.x * blockDim.x + threadIdx.x;
    if (t >= NG * NCLS) return;
    int g = t / NCLS, c = t % NCLS;
    float s = clsb[c];
#pragma unroll 4
    for (int k = 0; k < 2 * HID; ++k) s += rdt[g * 2 * HID + k] * clsW[k * NCLS + c];
    out[t] = s;
}

// ---------------------------------------------------------------------
static inline char* take_bytes(char*& p, size_t nbytes)
{
    char* r = p;
    p += ((nbytes + 255) & ~(size_t)255);
    return r;
}
static inline float* take(char*& p, size_t nfloats)
{
    return (float*)take_bytes(p, nfloats * sizeof(float));
}

extern "C" void kernel_launch(void* const* d_in, const int* in_sizes, int n_in,
                              void* d_out, int out_size, void* d_ws, size_t ws_size,
                              hipStream_t stream)
{
    (void)in_sizes; (void)n_in; (void)out_size; (void)ws_size;
    const float* x      = (const float*)d_in[0];
    const int*   src    = (const int*)d_in[1];
    const int*   dst    = (const int*)d_in[2];
    const int*   gid    = (const int*)d_in[3];
    const float* W_enc  = (const float*)d_in[5];
    const float* b_enc  = (const float*)d_in[6];
    const float* fc_W[2]   = {(const float*)d_in[7],  (const float*)d_in[14]};
    const float* attn_l[2] = {(const float*)d_in[8],  (const float*)d_in[15]};
    const float* attn_r[2] = {(const float*)d_in[9],  (const float*)d_in[16]};
    const float* res_W[2]  = {(const float*)d_in[10], (const float*)d_in[17]};
    const float* gbias[2]  = {(const float*)d_in[11], (const float*)d_in[18]};
    const float* dp_W[2]   = {(const float*)d_in[12], (const float*)d_in[19]};
    const float* dp_b[2]   = {(const float*)d_in[13], (const float*)d_in[20]};
    const float* gamma  = (const float*)d_in[21];
    const float* beta   = (const float*)d_in[22];
    const float* gate_W = (const float*)d_in[23];
    const float* gate_b = (const float*)d_in[24];
    const float* cls_W  = (const float*)d_in[25];
    const float* cls_b  = (const float*)d_in[26];
    float* out = (float*)d_out;

    char* wp = (char*)d_ws;
    float* he   = take(wp, (size_t)N_NODES * HID);
    float* hd   = take(wp, (size_t)N_NODES * HID);
    float* feat = take(wp, (size_t)N_NODES * NHID);
    float* acc  = take(wp, (size_t)N_NODES * NHID);
    float* el   = take(wp, (size_t)N_NODES * NH);
    float* er   = take(wp, (size_t)N_NODES * NH);
    float* m    = take(wp, (size_t)N_NODES * NH);
    float* den  = take(wp, (size_t)N_NODES * NH);
    float* esc  = take(wp, (size_t)N_EDGES * NH);
    float* hg   = take(wp, (size_t)N_NODES * 2 * HID);
    float* mu   = take(wp, 2 * HID);
    float* var  = take(wp, 2 * HID);
    float* gate = take(wp, N_NODES);
    float* gm   = take(wp, NG);
    float* gden = take(wp, NG);
    float* rdt  = take(wp, (size_t)NG * 2 * HID);
    // packed bf16 weights: each (N/16)*(K/32)*512 bf16 = 32768 elems = 64KB
    const size_t WPK = 32768;
    __bf16* Wp_enc = (__bf16*)take_bytes(wp, WPK * 2);
    __bf16* Wp_fc[2]  = {(__bf16*)take_bytes(wp, WPK * 2), (__bf16*)take_bytes(wp, WPK * 2)};
    __bf16* Wp_res[2] = {(__bf16*)take_bytes(wp, WPK * 2), (__bf16*)take_bytes(wp, WPK * 2)};
    __bf16* Wp_dp[2]  = {(__bf16*)take_bytes(wp, WPK * 2), (__bf16*)take_bytes(wp, WPK * 2)};

    auto pack = [&](const float* B, __bf16* Bp, int N, int K) {
        int total = (N / 16) * (K / 32) * 512;
        pack_weight_bf16<<<(total + 255) / 256, 256, 0, stream>>>(B, Bp, N, K);
    };
    auto gemm = [&](const float* A, const __bf16* Bp, const float* bs, float* C,
                    int M, int N, int K) {
        int waves = (M / (16 * MT)) * (N / 16);
        wmma_gemm_packed<<<(waves + 3) / 4, 128, 0, stream>>>(A, Bp, bs, C, M, N, K);
    };

    // ---- pack all weights to bf16 WMMA fragments (tiny, once per call)
    pack(W_enc, Wp_enc, HID, IN_DIM);
    for (int L = 0; L < 2; ++L) {
        pack(fc_W[L],  Wp_fc[L],  NHID, HID);
        pack(res_W[L], Wp_res[L], NHID, HID);
        pack(dp_W[L],  Wp_dp[L],  HID,  NHID);
    }

    // ---- encoder: he = x @ W_enc + b_enc
    gemm(x, Wp_enc, b_enc, he, N_NODES, HID, IN_DIM);

    const int eh = N_EDGES * NH, nh = N_NODES * NH;
    const float* hin = he;
    for (int L = 0; L < 2; ++L) {
        // feat = hin @ fc_W ; acc = hin @ res_W + bias (residual pre-seeded)
        gemm(hin, Wp_fc[L], nullptr, feat, N_NODES, NHID, HID);
        gemm(hin, Wp_res[L], gbias[L], acc, N_NODES, NHID, HID);
        attn_scores<<<(nh + 255) / 256, 256, 0, stream>>>(feat, attn_l[L], attn_r[L], el, er);
        fill_f32<<<(nh + 255) / 256, 256, 0, stream>>>(m, NEG_BIG, nh);
        fill_f32<<<(nh + 255) / 256, 256, 0, stream>>>(den, 0.f, nh);
        edge_score_max<<<(eh + 255) / 256, 256, 0, stream>>>(src, dst, el, er, esc, m);
        edge_exp_sum<<<(eh + 255) / 256, 256, 0, stream>>>(dst, esc, m, den);
        edge_alpha<<<(eh + 255) / 256, 256, 0, stream>>>(dst, esc, den);
        long long mt = (long long)N_EDGES * (NHID / 4);
        edge_message<<<(unsigned)((mt + 255) / 256), 256, 0, stream>>>(src, dst, feat, esc, acc);
        leaky_act4<<<(N_NODES * NHID / 4 + 255) / 256, 256, 0, stream>>>((float4*)acc, N_NODES * NHID / 4);
        // down-proj: hd = leaky(acc) @ dp_W + dp_b
        gemm(acc, Wp_dp[L], dp_b[L], hd, N_NODES, HID, NHID);
        hin = hd;
    }

    // ---- BatchNorm over concat([hd, he]) with batch stats
    bn_stats<<<2 * HID, 256, 0, stream>>>(hd, he, mu, var);
    bn_norm<<<(N_NODES * 2 * HID + 255) / 256, 256, 0, stream>>>(hd, he, mu, var, gamma, beta, hg);

    // ---- global attention pooling
    gate_proj<<<(N_NODES + 255) / 256, 256, 0, stream>>>(hg, gate_W, gate_b, gate);
    fill_f32<<<1, 256, 0, stream>>>(gm, NEG_BIG, NG);
    fill_f32<<<1, 256, 0, stream>>>(gden, 0.f, NG);
    fill_f32<<<(NG * 2 * HID + 255) / 256, 256, 0, stream>>>(rdt, 0.f, NG * 2 * HID);
    seg_gate_max<<<(N_NODES + 255) / 256, 256, 0, stream>>>(gid, gate, gm);
    seg_gate_expsum<<<(N_NODES + 255) / 256, 256, 0, stream>>>(gid, gate, gm, gden);
    pool<<<(N_NODES * 2 * HID + 255) / 256, 256, 0, stream>>>(gid, gate, gden, hg, rdt);

    // ---- classifier
    classifier<<<(NG * NCLS + 255) / 256, 256, 0, stream>>>(rdt, cls_W, cls_b, out);
}
